// jNzzEqqHtiqYqWsjrvuVQcwKYdkrmxdM_84980222919385
// MI455X (gfx1250) — compile-verified, matrix operand order checked by reading, not run on hardware
//
#include <hip/hip_runtime.h>
#include <math.h>

typedef unsigned short u16;
typedef __attribute__((ext_vector_type(16))) __bf16 v16bf;
typedef __attribute__((ext_vector_type(8)))  float  v8f;

// ---- CDNA5 async global->LDS copy support (probe via __has_builtin) ----
#if defined(__HIP_DEVICE_COMPILE__)
#if __has_builtin(__builtin_amdgcn_global_load_async_to_lds_b128)
#define HAVE_ASYNC_LDS 1
#endif
#endif
#ifndef HAVE_ASYNC_LDS
#define HAVE_ASYNC_LDS 0
#endif

#if HAVE_ASYNC_LDS
typedef int v4i __attribute__((vector_size(16)));
__device__ __forceinline__ void async_cp16(const u16* g, u16* l) {
    // builtin signature (from compiler diagnostic): global v4i* src, lds v4i* dst, imm, imm
    __builtin_amdgcn_global_load_async_to_lds_b128(
        (__attribute__((address_space(1))) v4i*)(void*)g,
        (__attribute__((address_space(3))) v4i*)(void*)l, 0, 0);
}
#if __has_builtin(__builtin_amdgcn_s_wait_asynccnt)
#define WAIT_ASYNC(n) __builtin_amdgcn_s_wait_asynccnt(n)
#else
#define WAIT_ASYNC(n) asm volatile("s_wait_asynccnt " #n ::: "memory")
#endif
#endif

__device__ __forceinline__ u16 f2bf(float f) {
    union { float f; unsigned u; } x; x.f = f;
    unsigned u = x.u;
    u += 0x7FFFu + ((u >> 16) & 1u);   // round to nearest even
    return (u16)(u >> 16);
}

// ---------------- elementwise f32 -> bf16 ----------------
__global__ void k_cvt_bf16(const float* __restrict__ in, u16* __restrict__ out, long n) {
    long i = (long)blockIdx.x * blockDim.x + threadIdx.x;
    long stride = (long)gridDim.x * blockDim.x;
    for (; i < n; i += stride) out[i] = f2bf(in[i]);
}

// ---------------- f32 [R][C] -> bf16 transposed [C][R] ----------------
__global__ void k_cvt_bf16_T(const float* __restrict__ in, u16* __restrict__ out,
                             int R, int C) {
    long i = (long)blockIdx.x * blockDim.x + threadIdx.x;
    long stride = (long)gridDim.x * blockDim.x;
    long n = (long)R * C;
    for (; i < n; i += stride) {
        int r = (int)(i / C), c = (int)(i % C);
        out[(long)c * R + r] = f2bf(in[i]);
    }
}

// ---------------- LayerNorm (row per block), bf16 out ----------------
__global__ void k_layernorm(const float* __restrict__ x, const float* __restrict__ w,
                            const float* __restrict__ b, u16* __restrict__ out, int D) {
    __shared__ float red[256];
    const int tid = threadIdx.x;
    const long row = blockIdx.x;
    const float* xr = x + row * (long)D;
    float s = 0.f, s2 = 0.f;
    for (int j = tid; j < D; j += 256) { float v = xr[j]; s += v; s2 += v * v; }
    red[tid] = s; __syncthreads();
    for (int o = 128; o > 0; o >>= 1) { if (tid < o) red[tid] += red[tid + o]; __syncthreads(); }
    float mean = red[0] / (float)D; __syncthreads();
    red[tid] = s2; __syncthreads();
    for (int o = 128; o > 0; o >>= 1) { if (tid < o) red[tid] += red[tid + o]; __syncthreads(); }
    float var = red[0] / (float)D - mean * mean;
    float rstd = rsqrtf(var + 1e-5f);
    for (int j = tid; j < D; j += 256)
        out[row * (long)D + j] = f2bf((xr[j] - mean) * rstd * w[j] + b[j]);
}

// -------- scale + softmax (row per block), bf16 probs out, padded out stride --------
__global__ void k_softmax_bf16(const float* __restrict__ S, int inStride,
                               u16* __restrict__ P, int outStride,
                               int L, float scale) {
    __shared__ float red[256];
    const int tid = threadIdx.x;
    const long row = blockIdx.x;
    const float* sr = S + row * (long)inStride;
    float m = -3.402823466e38f;
    for (int j = tid; j < L; j += 256) m = fmaxf(m, sr[j] * scale);
    red[tid] = m; __syncthreads();
    for (int o = 128; o > 0; o >>= 1) { if (tid < o) red[tid] = fmaxf(red[tid], red[tid + o]); __syncthreads(); }
    m = red[0]; __syncthreads();
    float s = 0.f;
    for (int j = tid; j < L; j += 256) s += __expf(sr[j] * scale - m);
    red[tid] = s; __syncthreads();
    for (int o = 128; o > 0; o >>= 1) { if (tid < o) red[tid] += red[tid + o]; __syncthreads(); }
    float inv = 1.f / red[0];
    for (int j = tid; j < L; j += 256)
        P[row * (long)outStride + j] = f2bf(__expf(sr[j] * scale - m) * inv);
}

// ---------------- bias + residual ----------------
__global__ void k_bias_res(const float* __restrict__ g, const float* __restrict__ bias,
                           const float* __restrict__ res, float* __restrict__ out,
                           int D, long n) {
    long i = (long)blockIdx.x * blockDim.x + threadIdx.x;
    long stride = (long)gridDim.x * blockDim.x;
    for (; i < n; i += stride) out[i] = g[i] + bias[i % D] + res[i];
}

// ---------------- GEGLU: a * gelu_exact(g), bf16 out ----------------
__global__ void k_geglu(const float* __restrict__ proj, const float* __restrict__ b1,
                        u16* __restrict__ out, int FFd, long rows) {
    long i = (long)blockIdx.x * blockDim.x + threadIdx.x;
    long stride = (long)gridDim.x * blockDim.x;
    long n = rows * (long)FFd;
    for (; i < n; i += stride) {
        long r = i / FFd; int j = (int)(i % FFd);
        float a = proj[r * (long)(2 * FFd) + j] + b1[j];
        float g = proj[r * (long)(2 * FFd) + FFd + j] + b1[FFd + j];
        float gel = 0.5f * g * (1.f + erff(g * 0.70710678118654752f));
        out[i] = f2bf(a * gel);
    }
}

// ================= bf16 WMMA GEMM: C = A @ Bt^T =================
// A: [M][K] row-major (lda), Bt: [N][K] row-major (ldb)  => C[m][n] = sum_k A[m,k]*Bt[n,k]
// Block: 256 thr = 8 waves; block tile 128(M) x 64(N), K-step 32; waves 4x2, each wave
// computes 32x32 (2 A-frags x 2 B-frags = 4 WMMA / K-step).
// Fast K-steps use double-buffered LDS: async global->LDS (ASYNCcnt) when available,
// otherwise register-prefetch + ds_store_b128.  K-tail / edge tiles take guarded path.
// OUT_MODE: 0 = f32, 1 = bf16, 2 = bf16 transposed (Cb[n*ldc + m])
template <int OUT_MODE>
__global__ __launch_bounds__(256) void k_gemm_bt(
    const u16* __restrict__ A, int lda,
    const u16* __restrict__ Bt, int ldb,
    void* __restrict__ C, int ldc,
    int M, int N, int K)
{
    __shared__ alignas(16) u16 As[2][128][40];   // [buf][m][k], 80B row stride
    __shared__ alignas(16) u16 Bs[2][64][40];    // [buf][n][k]

    const int tid  = threadIdx.x;
    const int lane = tid & 31;
    const int wid  = tid >> 5;
    const int wr   = wid & 3;        // wave row (M, 32-wide)
    const int wc   = wid >> 2;       // wave col (N, 32-wide)
    const int half = lane >> 4;
    const int r16  = lane & 15;
    const int blockM = blockIdx.y * 128;
    const int blockN = blockIdx.x * 64;

    const bool vecA = ((lda & 7) == 0) && (blockM + 128 <= M);
    const bool vecB = ((ldb & 7) == 0) && (blockN + 64 <= N);

    v8f acc00 = {}, acc01 = {}, acc10 = {}, acc11 = {};

    // per-thread tile-load coordinates (vector path)
    const int arow = tid >> 2, aseg = tid & 3;   // rows arow, arow+64; 8 bf16 per seg
    const int brow = tid >> 2, bseg = tid & 3;

    auto compute = [&](int buf) {
        union FU { uint4 q[2]; v16bf v; };
        FU a0, a1, b0, b1;
        const int am0 = wr * 32 + r16, am1 = am0 + 16;
        a0.q[0] = *reinterpret_cast<const uint4*>(&As[buf][am0][half * 8]);
        a0.q[1] = *reinterpret_cast<const uint4*>(&As[buf][am0][16 + half * 8]);
        a1.q[0] = *reinterpret_cast<const uint4*>(&As[buf][am1][half * 8]);
        a1.q[1] = *reinterpret_cast<const uint4*>(&As[buf][am1][16 + half * 8]);
        const int bn0 = wc * 32 + r16, bn1 = bn0 + 16;
        b0.q[0] = *reinterpret_cast<const uint4*>(&Bs[buf][bn0][half * 8]);
        b0.q[1] = *reinterpret_cast<const uint4*>(&Bs[buf][bn0][16 + half * 8]);
        b1.q[0] = *reinterpret_cast<const uint4*>(&Bs[buf][bn1][half * 8]);
        b1.q[1] = *reinterpret_cast<const uint4*>(&Bs[buf][bn1][16 + half * 8]);
        acc00 = __builtin_amdgcn_wmma_f32_16x16x32_bf16(false, a0.v, false, b0.v, (short)0, acc00, false, false);
        acc01 = __builtin_amdgcn_wmma_f32_16x16x32_bf16(false, a0.v, false, b1.v, (short)0, acc01, false, false);
        acc10 = __builtin_amdgcn_wmma_f32_16x16x32_bf16(false, a1.v, false, b0.v, (short)0, acc10, false, false);
        acc11 = __builtin_amdgcn_wmma_f32_16x16x32_bf16(false, a1.v, false, b1.v, (short)0, acc11, false, false);
    };

    const int numFast = (vecA && vecB) ? (K >> 5) : 0;

    if (numFast > 0) {
#if HAVE_ASYNC_LDS
        auto issue = [&](int s) {
            const int buf = s & 1, k0 = s << 5;
            async_cp16(A  + (long)(blockM + arow)      * lda + k0 + aseg * 8, &As[buf][arow][aseg * 8]);
            async_cp16(A  + (long)(blockM + arow + 64) * lda + k0 + aseg * 8, &As[buf][arow + 64][aseg * 8]);
            async_cp16(Bt + (long)(blockN + brow)      * ldb + k0 + bseg * 8, &Bs[buf][brow][bseg * 8]);
        };
        issue(0);
        for (int s = 0; s < numFast; ++s) {
            if (s + 1 < numFast) { issue(s + 1); WAIT_ASYNC(3); }
            else                 { WAIT_ASYNC(0); }
            __syncthreads();                 // tile s fully in LDS for all waves
            compute(s & 1);
            __syncthreads();                 // done reading buf before it is refilled
        }
#else
        uint4 ra0, ra1, rb;
        auto gload = [&](int s) {
            const int k0 = s << 5;
            ra0 = *reinterpret_cast<const uint4*>(A  + (long)(blockM + arow)      * lda + k0 + aseg * 8);
            ra1 = *reinterpret_cast<const uint4*>(A  + (long)(blockM + arow + 64) * lda + k0 + aseg * 8);
            rb  = *reinterpret_cast<const uint4*>(Bt + (long)(blockN + brow)      * ldb + k0 + bseg * 8);
        };
        gload(0);
        for (int s = 0; s < numFast; ++s) {
            const int buf = s & 1;
            __syncthreads();                 // buf free (compute on same buf finished)
            *reinterpret_cast<uint4*>(&As[buf][arow][aseg * 8])      = ra0;
            *reinterpret_cast<uint4*>(&As[buf][arow + 64][aseg * 8]) = ra1;
            *reinterpret_cast<uint4*>(&Bs[buf][brow][bseg * 8])      = rb;
            if (s + 1 < numFast) gload(s + 1);   // overlap next loads with compute
            __syncthreads();
            compute(buf);
        }
#endif
    }

    // guarded K-tail (and full guarded loop when the block touches an edge)
    const int kStart = numFast << 5;
    for (int k0 = kStart; k0 < K; k0 += 32) {
        __syncthreads();
#pragma unroll
        for (int i = 0; i < 16; ++i) {
            int idx = tid + i * 256;
            int row = idx >> 5, cc = idx & 31;
            int gm = blockM + row, gk = k0 + cc;
            As[0][row][cc] = (gm < M && gk < K) ? A[(long)gm * lda + gk] : (u16)0;
        }
#pragma unroll
        for (int i = 0; i < 8; ++i) {
            int idx = tid + i * 256;
            int row = idx >> 5, kk = idx & 31;
            int gn = blockN + row, gk = k0 + kk;
            Bs[0][row][kk] = (gn < N && gk < K) ? Bt[(long)gn * ldb + gk] : (u16)0;
        }
        __syncthreads();
        compute(0);
    }

    // C/D layout: VGPR r, lane -> M = r + 8*half, N = lane%16
#pragma unroll
    for (int ar = 0; ar < 2; ++ar) {
        const v8f* accs[2] = { ar ? &acc10 : &acc00, ar ? &acc11 : &acc01 };
#pragma unroll
        for (int r = 0; r < 8; ++r) {
            int m = blockM + wr * 32 + ar * 16 + r + 8 * half;
            if (m >= M) continue;
#pragma unroll
            for (int bc = 0; bc < 2; ++bc) {
                int n = blockN + wc * 32 + bc * 16 + r16;
                if (n >= N) continue;
                float v = (*accs[bc])[r];
                if (OUT_MODE == 0)      ((float*)C)[(long)m * ldc + n] = v;
                else if (OUT_MODE == 1) ((u16*)C)[(long)m * ldc + n] = f2bf(v);
                else                    ((u16*)C)[(long)n * ldc + m] = f2bf(v);  // transposed
            }
        }
    }
}

// ---------------- host-side helper ----------------
static void gemm(int outMode, const u16* A, int lda, const u16* Bt, int ldb,
                 void* C, int ldc, int M, int N, int K, hipStream_t s) {
    dim3 grid((N + 63) / 64, (M + 127) / 128);
    dim3 blk(256);
    if (outMode == 0)      k_gemm_bt<0><<<grid, blk, 0, s>>>(A, lda, Bt, ldb, C, ldc, M, N, K);
    else if (outMode == 1) k_gemm_bt<1><<<grid, blk, 0, s>>>(A, lda, Bt, ldb, C, ldc, M, N, K);
    else                   k_gemm_bt<2><<<grid, blk, 0, s>>>(A, lda, Bt, ldb, C, ldc, M, N, K);
}

extern "C" void kernel_launch(void* const* d_in, const int* in_sizes, int n_in,
                              void* d_out, int out_size, void* d_ws, size_t ws_size,
                              hipStream_t stream) {
    (void)in_sizes; (void)n_in; (void)out_size; (void)ws_size;

    const int Bb = 2, Nt = 4096, Dm = 640, Hh = 8, DHd = 80;
    const int CNt = 77, CDd = 768, FFd = 2560;
    const long tokens = (long)Bb * Nt;            // 8192
    const float scl = 1.0f / sqrtf((float)DHd);

    const float* x       = (const float*)d_in[0];
    const float* context = (const float*)d_in[1];
    const float* ln1_w = (const float*)d_in[2],  *ln1_b = (const float*)d_in[3];
    const float* ln2_w = (const float*)d_in[4],  *ln2_b = (const float*)d_in[5];
    const float* ln3_w = (const float*)d_in[6],  *ln3_b = (const float*)d_in[7];
    const float* wq1 = (const float*)d_in[8],  *wk1 = (const float*)d_in[9];
    const float* wv1 = (const float*)d_in[10], *wo1 = (const float*)d_in[11];
    const float* bo1 = (const float*)d_in[12];
    const float* wq2 = (const float*)d_in[13], *wk2 = (const float*)d_in[14];
    const float* wv2 = (const float*)d_in[15], *wo2 = (const float*)d_in[16];
    const float* bo2 = (const float*)d_in[17];
    const float* w_ff1 = (const float*)d_in[18], *b_ff1 = (const float*)d_in[19];
    const float* w_ff2 = (const float*)d_in[20], *b_ff2 = (const float*)d_in[21];

    // ---- workspace carve ----
    char* p = (char*)d_ws;
    auto take = [&](size_t bytes) -> void* {
        void* r = (void*)p;
        p += (bytes + 255) & ~(size_t)255;
        return r;
    };
    // transposed bf16 weights: wXt[n][k]
    u16* wq1t = (u16*)take((size_t)Dm * Dm * 2);
    u16* wk1t = (u16*)take((size_t)Dm * Dm * 2);
    u16* wv1t = (u16*)take((size_t)Dm * Dm * 2);
    u16* wo1t = (u16*)take((size_t)Dm * Dm * 2);
    u16* wq2t = (u16*)take((size_t)Dm * Dm * 2);
    u16* wk2t = (u16*)take((size_t)CDd * Dm * 2);
    u16* wv2t = (u16*)take((size_t)CDd * Dm * 2);
    u16* wo2t = (u16*)take((size_t)Dm * Dm * 2);
    u16* wf1t = (u16*)take((size_t)Dm * 2 * FFd * 2);
    u16* wf2t = (u16*)take((size_t)FFd * Dm * 2);
    u16* ctxb = (u16*)take((size_t)Bb * CNt * CDd * 2);
    u16* h_bf = (u16*)take((size_t)tokens * Dm * 2);
    u16* q_bf = (u16*)take((size_t)tokens * Dm * 2);
    u16* k_bf = (u16*)take((size_t)tokens * Dm * 2);
    u16* vT   = (u16*)take((size_t)Dm * tokens * 2);    // self V transposed [d][token]
    u16* vT2  = (u16*)take((size_t)Dm * 160 * 2);       // cross V transposed [d][b*80+t]
    u16* o_bf = (u16*)take((size_t)tokens * Dm * 2);
    u16* p_bf = (u16*)take((size_t)Nt * Nt * 2);        // probs scratch (reused per b,h)
    u16* ffin = (u16*)take((size_t)tokens * FFd * 2);
    float* scoresf = (float*)take((size_t)Nt * Nt * 4); // scores scratch (reused per b,h)
    float* tmpf    = (float*)take((size_t)tokens * 2 * FFd * 4);
    float* x1      = (float*)take((size_t)tokens * Dm * 4);
    float* x2      = (float*)take((size_t)tokens * Dm * 4);

    // ---- weight conversion (transposed) + context conversion ----
    struct { const float* src; u16* dst; int R, C; } wt[10] = {
        { wq1, wq1t, Dm, Dm }, { wk1, wk1t, Dm, Dm }, { wv1, wv1t, Dm, Dm },
        { wo1, wo1t, Dm, Dm }, { wq2, wq2t, Dm, Dm }, { wk2, wk2t, CDd, Dm },
        { wv2, wv2t, CDd, Dm }, { wo2, wo2t, Dm, Dm },
        { w_ff1, wf1t, Dm, 2 * FFd }, { w_ff2, wf2t, FFd, Dm },
    };
    for (int i = 0; i < 10; ++i)
        k_cvt_bf16_T<<<1024, 256, 0, stream>>>(wt[i].src, wt[i].dst, wt[i].R, wt[i].C);
    k_cvt_bf16<<<256, 256, 0, stream>>>(context, ctxb, (long)Bb * CNt * CDd);

    // ================= self-attention =================
    k_layernorm<<<(int)tokens, 256, 0, stream>>>(x, ln1_w, ln1_b, h_bf, Dm);
    gemm(1, h_bf, Dm, wq1t, Dm, q_bf, Dm, (int)tokens, Dm, Dm, stream);
    gemm(1, h_bf, Dm, wk1t, Dm, k_bf, Dm, (int)tokens, Dm, Dm, stream);
    gemm(2, h_bf, Dm, wv1t, Dm, vT, (int)tokens, (int)tokens, Dm, Dm, stream);  // vT[d][tok]
    for (int z = 0; z < Bb * Hh; ++z) {
        int b = z / Hh, hh = z % Hh;
        const u16* Aq = q_bf + (long)b * Nt * Dm + hh * DHd;
        const u16* Bk = k_bf + (long)b * Nt * Dm + hh * DHd;   // already [n=key][k=dh]
        gemm(0, Aq, Dm, Bk, Dm, scoresf, Nt, Nt, Nt, DHd, stream);
        k_softmax_bf16<<<Nt, 256, 0, stream>>>(scoresf, Nt, p_bf, Nt, Nt, scl);
        const u16* Bv = vT + (long)hh * DHd * tokens + (long)b * Nt;  // [n=d][k=tok]
        u16* Co = o_bf + (long)b * Nt * Dm + hh * DHd;
        gemm(1, p_bf, Nt, Bv, (int)tokens, Co, Dm, Nt, DHd, Nt, stream);
    }
    gemm(0, o_bf, Dm, wo1t, Dm, tmpf, Dm, (int)tokens, Dm, Dm, stream);
    k_bias_res<<<2048, 256, 0, stream>>>(tmpf, bo1, x, x1, Dm, tokens * Dm);

    // ================= cross-attention =================
    k_layernorm<<<(int)tokens, 256, 0, stream>>>(x1, ln2_w, ln2_b, h_bf, Dm);
    gemm(1, h_bf, Dm, wq2t, Dm, q_bf, Dm, (int)tokens, Dm, Dm, stream);
    gemm(1, ctxb, CDd, wk2t, CDd, k_bf, Dm, Bb * CNt, Dm, CDd, stream);
    for (int b = 0; b < Bb; ++b)   // per-batch transposed V: vT2[d][b*80 + t], ld 160
        gemm(2, ctxb + (long)b * CNt * CDd, CDd, wv2t, CDd,
             vT2 + b * 80, 160, CNt, Dm, CDd, stream);
    for (int z = 0; z < Bb * Hh; ++z) {
        int b = z / Hh, hh = z % Hh;
        const u16* Aq = q_bf + (long)b * Nt * Dm + hh * DHd;
        const u16* Bk = k_bf + (long)b * CNt * Dm + hh * DHd;
        gemm(0, Aq, Dm, Bk, Dm, scoresf, CNt, Nt, CNt, DHd, stream);
        k_softmax_bf16<<<Nt, 256, 0, stream>>>(scoresf, CNt, p_bf, 80, CNt, scl);
        const u16* Bv = vT2 + (long)hh * DHd * 160 + b * 80;   // [n=d][k=ctx tok]
        u16* Co = o_bf + (long)b * Nt * Dm + hh * DHd;
        gemm(1, p_bf, 80, Bv, 160, Co, Dm, Nt, DHd, CNt, stream);
    }
    gemm(0, o_bf, Dm, wo2t, Dm, tmpf, Dm, (int)tokens, Dm, Dm, stream);
    k_bias_res<<<2048, 256, 0, stream>>>(tmpf, bo2, x1, x2, Dm, tokens * Dm);

    // ================= GEGLU feed-forward =================
    k_layernorm<<<(int)tokens, 256, 0, stream>>>(x2, ln3_w, ln3_b, h_bf, Dm);
    gemm(0, h_bf, Dm, wf1t, Dm, tmpf, 2 * FFd, (int)tokens, 2 * FFd, Dm, stream);
    k_geglu<<<4096, 256, 0, stream>>>(tmpf, b_ff1, ffin, FFd, tokens);
    gemm(0, ffin, FFd, wf2t, FFd, tmpf, Dm, (int)tokens, Dm, FFd, stream);
    k_bias_res<<<2048, 256, 0, stream>>>(tmpf, b_ff2, x2, (float*)d_out, Dm, tokens * Dm);
}
